// TriangleMultiplicationOutgoing_50173807952273
// MI455X (gfx1250) — matrix operand order repaired and checked by reading, not compiled, not run on hardware
//
#include <hip/hip_runtime.h>
#include <hip/hip_bf16.h>

// MI455X / gfx1250: all GEMM work on v_wmma_f32_16x16x32_f16 (f16 in, f32 acc).
// a/b intermediates f16 channel-major (67 MB each -> L2-resident, 192 MB L2).
// t stored f32 channel-major for coalesced stores; epilogue transposes via LDS.

typedef __attribute__((ext_vector_type(16))) _Float16 v16h;
typedef __attribute__((ext_vector_type(8)))  _Float16 v8h;
typedef __attribute__((ext_vector_type(4)))  _Float16 v4h;
typedef __attribute__((ext_vector_type(8)))  float    v8f;
typedef __attribute__((ext_vector_type(4)))  float    v4f;

#define NN 512
#define DD 128
#define RR (NN * NN)           // 262144 rows
#define LN_EPS 1e-5f

__device__ __forceinline__ v8f wmma16(v16h a, v16h b, v8f c) {
    return __builtin_amdgcn_wmma_f32_16x16x32_f16(false, a, false, b, (short)0, c,
                                                  false, false);
}

__device__ __forceinline__ v16h ld16h(const _Float16* p0, const _Float16* p1) {
    union { v16h v; v8h h[2]; } u;
    u.h[0] = *(const v8h*)p0;
    u.h[1] = *(const v8h*)(p1);
    return u.v;
}

// fast sigmoid: v_exp_f32 + v_add + v_rcp_f32 (no precise-division expansion)
__device__ __forceinline__ float sigmoidf_(float v) {
    return __builtin_amdgcn_rcpf(1.0f + __expf(-v));
}

// LDS tile index with XOR swizzle on bit3 of the major index: the two
// half-wave lane groups (which differ by 8 in the major index) land on
// disjoint bank groups.  Flipping col-bit4 preserves 4-float contiguity.
__device__ __forceinline__ int swz(int major, int col) {
    return major * 128 + (col ^ (((major >> 3) & 1) << 4));
}

// ---------------------------------------------------------------------------
// Kernel 0: weights -> f16, transposed to [out][in] (contiguous B fragments).
// ---------------------------------------------------------------------------
__global__ __launch_bounds__(256)
void prep_weights(const float* __restrict__ wpin,  // [128][256]
                  const float* __restrict__ wgin,  // [128][256]
                  const float* __restrict__ wpout, // [128][128]
                  const float* __restrict__ wgout, // [128][128]
                  _Float16* __restrict__ tpin,     // [256][128]
                  _Float16* __restrict__ tgin,     // [256][128]
                  _Float16* __restrict__ tpout,    // [128][128]
                  _Float16* __restrict__ tgout)    // [128][128]
{
    const int idx = blockIdx.x * 256 + threadIdx.x;   // 0 .. 32767
    const int o = idx >> 7;
    const int i = idx & 127;
    tpin[idx] = (_Float16)wpin[i * 256 + o];
    tgin[idx] = (_Float16)wgin[i * 256 + o];
    if (idx < 128 * 128) {
        tpout[idx] = (_Float16)wpout[i * 128 + o];
        tgout[idx] = (_Float16)wgout[i * 128 + o];
    }
}

// ---------------------------------------------------------------------------
// Kernel 1: prologue.  layernorm(x) -> WMMA p/g/gate -> h = p*sig(g)*mask,
// a|b stored channel-major f16 [d][row]; gate bounced through LDS so the
// [row][d] f16 store is coalesced.
// ---------------------------------------------------------------------------
__global__ __launch_bounds__(256)
void prologue_kernel(const float* __restrict__ x,
                     const float* __restrict__ mask,
                     const float* __restrict__ ninw,
                     const float* __restrict__ ninb,
                     const _Float16* __restrict__ tpin,
                     const _Float16* __restrict__ tgin,
                     const _Float16* __restrict__ tgout,
                     _Float16* __restrict__ aF,    // [128][262144]
                     _Float16* __restrict__ bF,    // [128][262144]
                     _Float16* __restrict__ gate)  // [262144][128]
{
    __shared__ float gl[128 * 128];                 // 64 KB gate tile
    const int tid  = threadIdx.x;
    const int lane = tid & 31;
    const int wv   = tid >> 5;
    const int ln   = lane & 15;
    const int hi   = lane >> 4;
    const int rblk = blockIdx.x * 128;
    const int r0   = rblk + wv * 16;
    const int row  = r0 + ln;

    // ---- LayerNorm (lane covers 64 of 128 channels of its row)
    float xv[64];
    const float* xr = x + (size_t)row * DD;
    float s = 0.f, ss = 0.f;
#pragma unroll
    for (int c = 0; c < 8; ++c) {
        const int k0 = c * 16 + hi * 8;
        v4f a0 = *(const v4f*)(xr + k0);
        v4f a1 = *(const v4f*)(xr + k0 + 4);
#pragma unroll
        for (int e = 0; e < 4; ++e) {
            xv[c * 8 + e]     = a0[e];
            xv[c * 8 + 4 + e] = a1[e];
            s  += a0[e] + a1[e];
            ss += a0[e] * a0[e] + a1[e] * a1[e];
        }
    }
    s  += __shfl_xor(s, 16, 32);
    ss += __shfl_xor(ss, 16, 32);
    const float mu  = s * (1.0f / DD);
    const float var = ss * (1.0f / DD) - mu * mu;
    const float rs  = rsqrtf(var + LN_EPS);

    // ---- A fragments (16x32 f16, K=0..127)
    v16h af[4];
#pragma unroll
    for (int c = 0; c < 8; ++c) {
        const int k0 = c * 16 + hi * 8;
        v4f w0 = *(const v4f*)(ninw + k0);
        v4f w1 = *(const v4f*)(ninw + k0 + 4);
        v4f b0 = *(const v4f*)(ninb + k0);
        v4f b1 = *(const v4f*)(ninb + k0 + 4);
#pragma unroll
        for (int e = 0; e < 4; ++e) {
            af[c >> 1][(c & 1) * 8 + e]     = (_Float16)((xv[c*8+e]   - mu) * rs * w0[e] + b0[e]);
            af[c >> 1][(c & 1) * 8 + 4 + e] = (_Float16)((xv[c*8+4+e] - mu) * rs * w1[e] + b1[e]);
        }
    }

    float mk[8];
    const float* mp = mask + r0 + hi * 8;
#pragma unroll
    for (int e = 0; e < 8; ++e) mk[e] = mp[e];

    // ---- p / g projections (256 outputs = 16 column blocks)
    for (int ob = 0; ob < 16; ++ob) {
        const int o = ob * 16 + ln;
        v8f accP = {};
        v8f accG = {};
#pragma unroll
        for (int f = 0; f < 4; ++f) {
            const _Float16* wp = tpin + o * DD + f * 32 + hi * 8;
            const _Float16* wg = tgin + o * DD + f * 32 + hi * 8;
            accP = wmma16(af[f], ld16h(wp, wp + 16), accP);
            accG = wmma16(af[f], ld16h(wg, wg + 16), accG);
        }
        union { v8h h; _Float16 e[8]; } hv;
#pragma unroll
        for (int e = 0; e < 8; ++e)
            hv.e[e] = (_Float16)(accP[e] * sigmoidf_(accG[e]) * mk[e]);
        _Float16* dst = (ob < 8 ? aF : bF)
                      + (size_t)((ob & 7) * 16 + ln) * RR + r0 + hi * 8;
        *(v8h*)dst = hv.h;
    }

    // ---- output gate -> LDS tile (swizzled), then coalesced f16 store
    for (int ob = 0; ob < 8; ++ob) {
        const int o = ob * 16 + ln;
        v8f acc = {};
#pragma unroll
        for (int f = 0; f < 4; ++f) {
            const _Float16* wg = tgout + o * DD + f * 32 + hi * 8;
            acc = wmma16(af[f], ld16h(wg, wg + 16), acc);
        }
#pragma unroll
        for (int e = 0; e < 8; ++e)
            gl[swz(wv * 16 + hi * 8 + e, o)] = sigmoidf_(acc[e]);
    }
    __syncthreads();
#pragma unroll
    for (int it = 0; it < 16; ++it) {
        const int idx4 = it * 256 + tid;       // 0..4095 quads
        const int rw   = idx4 >> 5;            // local row 0..127
        const int c4   = (idx4 & 31) * 4;      // channel quad
        v4h h;
#pragma unroll
        for (int k = 0; k < 4; ++k) h[k] = (_Float16)gl[swz(rw, c4 + k)];
        *(v4h*)(gate + (size_t)(rblk + rw) * DD + c4) = h;
    }
}

// ---------------------------------------------------------------------------
// Kernel 2: triangle einsum.  t[d] = A_d (512x512) @ B_d^T, per-wave 64x64
// tile, 4x4 accumulators, 16 K-steps.  Stores coalesced f32 [d][i][j].
// ---------------------------------------------------------------------------
__global__ __launch_bounds__(256)
void triangle_kernel(const _Float16* __restrict__ aF,
                     const _Float16* __restrict__ bF,
                     float* __restrict__ tF)        // [128][512][512]
{
    const int lane = threadIdx.x & 31;
    const int wv   = threadIdx.x >> 5;
    const int ln   = lane & 15;
    const int hi   = lane >> 4;
    const int d    = blockIdx.x;
    const int i0   = blockIdx.y * 128 + (wv >> 2) * 64;
    const int j0   = blockIdx.z * 256 + (wv & 3) * 64;

    const _Float16* aB = aF + (size_t)d * RR;
    const _Float16* bB = bF + (size_t)d * RR;

    v8f acc[4][4];
#pragma unroll
    for (int ti = 0; ti < 4; ++ti)
#pragma unroll
        for (int tj = 0; tj < 4; ++tj) acc[ti][tj] = (v8f){};

    for (int kb = 0; kb < NN; kb += 32) {
        v16h a[4], b[4];
#pragma unroll
        for (int t = 0; t < 4; ++t) {
            const _Float16* ap = aB + (size_t)(i0 + t * 16 + ln) * NN + kb + hi * 8;
            a[t] = ld16h(ap, ap + 16);
            const _Float16* bp = bB + (size_t)(j0 + t * 16 + ln) * NN + kb + hi * 8;
            b[t] = ld16h(bp, bp + 16);
        }
        if (kb + 32 < NN) {           // global_prefetch_b8 toward next K-block
            __builtin_prefetch(aB + (size_t)(i0 + ln) * NN + kb + 32, 0, 1);
            __builtin_prefetch(bB + (size_t)(j0 + ln) * NN + kb + 32, 0, 1);
        }
#pragma unroll
        for (int ti = 0; ti < 4; ++ti)
#pragma unroll
            for (int tj = 0; tj < 4; ++tj)
                acc[ti][tj] = wmma16(a[ti], b[tj], acc[ti][tj]);
    }

    float* tB = tF + (size_t)d * RR;
#pragma unroll
    for (int ti = 0; ti < 4; ++ti)
#pragma unroll
        for (int tj = 0; tj < 4; ++tj)
#pragma unroll
            for (int e = 0; e < 8; ++e)   // 16-lane contiguous f32 stores
                tB[(size_t)(i0 + ti * 16 + hi * 8 + e) * NN + j0 + tj * 16 + ln]
                    = acc[ti][tj][e];
}

// ---------------------------------------------------------------------------
// Kernel 3: epilogue.  LDS-transpose a [128 d][128 j] tile of t, layernorm
// over d, WMMA with w_p_out, bounce result through LDS so out-store and
// gate-load are coalesced.
// ---------------------------------------------------------------------------
__global__ __launch_bounds__(256)
void epilogue_kernel(const float* __restrict__ tF,       // [128][512][512]
                     const _Float16* __restrict__ tpout,
                     const _Float16* __restrict__ gate,  // [262144][128]
                     const float* __restrict__ now,
                     const float* __restrict__ nob,
                     float* __restrict__ out)            // [262144][128]
{
    __shared__ float tl[128 * 128];                      // 64 KB, reused twice
    const int tid  = threadIdx.x;
    const int lane = tid & 31;
    const int wv   = tid >> 5;
    const int ln   = lane & 15;
    const int hi   = lane >> 4;
    const int r0   = blockIdx.x * 128;                   // 128 rows, same i

    // ---- phase 1: coalesced load of t[d][r0..r0+127] -> LDS (d-major, swizzled)
#pragma unroll
    for (int it = 0; it < 16; ++it) {
        const int idx4 = it * 256 + tid;                 // 0..4095 quads
        const int d    = idx4 >> 5;
        const int jj   = (idx4 & 31) * 4;
        v4f v = *(const v4f*)(tF + (size_t)d * RR + r0 + jj);
        *(v4f*)(&tl[swz(d, jj)]) = v;
    }
    __syncthreads();

    // ---- phase 2: layernorm over d (transposed LDS reads) + WMMA projection
    const int jl = wv * 16 + ln;                         // local row
    float tv[64];
    float s = 0.f, ss = 0.f;
#pragma unroll
    for (int c = 0; c < 8; ++c) {
        const int k0 = c * 16 + hi * 8;
#pragma unroll
        for (int e = 0; e < 8; ++e) {
            float v = tl[swz(k0 + e, jl)];
            tv[c * 8 + e] = v;
            s += v;
            ss += v * v;
        }
    }
    s  += __shfl_xor(s, 16, 32);
    ss += __shfl_xor(ss, 16, 32);
    const float mu  = s * (1.0f / DD);
    const float var = ss * (1.0f / DD) - mu * mu;
    const float rs  = rsqrtf(var + LN_EPS);

    v16h af[4];
#pragma unroll
    for (int c = 0; c < 8; ++c) {
        const int k0 = c * 16 + hi * 8;
        v4f w0 = *(const v4f*)(now + k0);
        v4f w1 = *(const v4f*)(now + k0 + 4);
        v4f b0 = *(const v4f*)(nob + k0);
        v4f b1 = *(const v4f*)(nob + k0 + 4);
#pragma unroll
        for (int e = 0; e < 4; ++e) {
            af[c >> 1][(c & 1) * 8 + e]     = (_Float16)((tv[c*8+e]   - mu) * rs * w0[e] + b0[e]);
            af[c >> 1][(c & 1) * 8 + 4 + e] = (_Float16)((tv[c*8+4+e] - mu) * rs * w1[e] + b1[e]);
        }
    }

    v8f acc[8];
#pragma unroll
    for (int ob = 0; ob < 8; ++ob) {
        const int o = ob * 16 + ln;
        v8f a_ = {};
#pragma unroll
        for (int f = 0; f < 4; ++f) {
            const _Float16* wp = tpout + o * DD + f * 32 + hi * 8;
            a_ = wmma16(af[f], ld16h(wp, wp + 16), a_);
        }
        acc[ob] = a_;
    }
    __syncthreads();                 // done reading t-tile, reuse LDS for output

    // ---- phase 3: deposit result tile (row-major, swizzled)
#pragma unroll
    for (int ob = 0; ob < 8; ++ob)
#pragma unroll
        for (int e = 0; e < 8; ++e)
            tl[swz(wv * 16 + hi * 8 + e, ob * 16 + ln)] = acc[ob][e];
    __syncthreads();

    // ---- phase 4: coalesced gate-multiply + store
#pragma unroll
    for (int it = 0; it < 16; ++it) {
        const int idx4 = it * 256 + tid;
        const int rw   = idx4 >> 5;
        const int c4   = (idx4 & 31) * 4;
        const size_t g = (size_t)(r0 + rw) * DD + c4;
        v4h gh = *(const v4h*)(gate + g);
        v4f ov;
#pragma unroll
        for (int k = 0; k < 4; ++k) ov[k] = tl[swz(rw, c4 + k)] * (float)gh[k];
        *(v4f*)(out + g) = ov;
    }
}

// ---------------------------------------------------------------------------
// Host launcher
// ---------------------------------------------------------------------------
extern "C" void kernel_launch(void* const* d_in, const int* in_sizes, int n_in,
                              void* d_out, int out_size, void* d_ws, size_t ws_size,
                              hipStream_t stream) {
    const float* x       = (const float*)d_in[0];
    const float* mask    = (const float*)d_in[1];
    const float* w_p_in  = (const float*)d_in[2];
    const float* w_g_in  = (const float*)d_in[3];
    const float* w_p_out = (const float*)d_in[4];
    const float* w_g_out = (const float*)d_in[5];
    const float* ninw    = (const float*)d_in[6];
    const float* ninb    = (const float*)d_in[7];
    const float* now     = (const float*)d_in[8];
    const float* nob     = (const float*)d_in[9];
    float* out = (float*)d_out;

    // workspace carve-up: weights 0.2 MB + a 67 + b 67 + gate 67 + t(f32) 134 MB
    _Float16* tpin  = (_Float16*)d_ws;
    _Float16* tgin  = tpin  + 256 * 128;
    _Float16* tpout = tgin  + 256 * 128;
    _Float16* tgout = tpout + 128 * 128;
    _Float16* aF    = tgout + 128 * 128;
    _Float16* bF    = aF + (size_t)DD * RR;
    _Float16* gateW = bF + (size_t)DD * RR;
    float*    tFw   = (float*)(gateW + (size_t)RR * DD);
    (void)in_sizes; (void)n_in; (void)out_size; (void)ws_size;

    prep_weights<<<128, 256, 0, stream>>>(w_p_in, w_g_in, w_p_out, w_g_out,
                                          tpin, tgin, tpout, tgout);
    prologue_kernel<<<RR / 128, 256, 0, stream>>>(x, mask, ninw, ninb,
                                                  tpin, tgin, tgout,
                                                  aF, bF, gateW);
    triangle_kernel<<<dim3(DD, NN / 128, NN / 256), 256, 0, stream>>>(aF, bF, tFw);
    epilogue_kernel<<<RR / 128, 256, 0, stream>>>(tFw, tpout, gateW, now, nob, out);
}
